// SelfAttention_12171937317354
// MI455X (gfx1250) — compile-verified, matrix operand order checked by reading, not run on hardware
//
#include <hip/hip_runtime.h>
#include <hip/hip_bf16.h>

typedef _Float16 v16h __attribute__((ext_vector_type(16)));
typedef _Float16 v8h  __attribute__((ext_vector_type(8)));
typedef _Float16 v4h  __attribute__((ext_vector_type(4)));
typedef float    v8f  __attribute__((ext_vector_type(8)));

#define T_  128
#define C_  768
#define XP  776      // LDS pitch for X (halves), multiple of 8
#define SP  136      // LDS pitch for 128-wide tiles (halves), multiple of 8
#define WN  (768*768)

static __device__ __forceinline__ v8f wmma16(v16h a, v16h b, v8f c) {
    return __builtin_amdgcn_wmma_f32_16x16x32_f16(false, a, false, b, (short)0, c, false, false);
}

// Load a 16x32 f16 A (or 32x16 B) fragment slice for this lane.
// p points at (row-of-this-lane)*pitch + k0; h = lane/16.
// Lane holds K in [8h, 8h+8) and [16+8h, 16+8h+8)  (CDNA5 16-bit A/B VGPR layout).
static __device__ __forceinline__ v16h load_frag(const _Float16* p, int h) {
    v8h lo = *(const v8h*)(p + 8 * h);
    v8h hi = *(const v8h*)(p + 16 + 8 * h);
    return __builtin_shufflevector(lo, hi, 0,1,2,3,4,5,6,7,8,9,10,11,12,13,14,15);
}

__global__ __launch_bounds__(256)
void wcvt_kernel(const float* __restrict__ Wq, const float* __restrict__ Wk,
                 const float* __restrict__ Wv, _Float16* __restrict__ wh) {
    int i = blockIdx.x * blockDim.x + threadIdx.x;
    if (i < WN) {
        wh[i]          = (_Float16)Wq[i];
        wh[WN + i]     = (_Float16)Wk[i];
        wh[2 * WN + i] = (_Float16)Wv[i];
    }
}

__global__ __launch_bounds__(256, 2)
void attn_fused_kernel(const float* __restrict__ x,
                       const _Float16* __restrict__ WhQ,
                       const _Float16* __restrict__ WhK,
                       const _Float16* __restrict__ WhV,
                       const float* __restrict__ bq,
                       const float* __restrict__ bk,
                       const float* __restrict__ bv,
                       float* __restrict__ out) {
    __shared__ _Float16 Xs[T_ * XP];   // 198,656 B : X_b as f16
    __shared__ _Float16 Qs[T_ * SP];   //  34,816 B : Q chunk   (later aliased as P)
    __shared__ _Float16 Ks[T_ * SP];   //  34,816 B : K chunk   (later aliased as V^T)

    _Float16* Ps  = Qs;
    _Float16* VsT = Ks;

    const int tid  = threadIdx.x;
    const int w    = tid >> 5;        // wave id 0..7
    const int lane = tid & 31;
    const int idx  = lane & 15;
    const int h    = lane >> 4;
    const int b    = blockIdx.x;

    const float* xb = x + (size_t)b * (T_ * C_);

    // ---- Phase 0: stage X_b into LDS as f16 ----
    for (int p = tid; p < (T_ * C_) / 4; p += 256) {
        int r = p / (C_ / 4);
        int c = (p % (C_ / 4)) * 4;
        float4 vv = *(const float4*)(xb + (size_t)r * C_ + c);
        v4h o;
        o[0] = (_Float16)vv.x; o[1] = (_Float16)vv.y;
        o[2] = (_Float16)vv.z; o[3] = (_Float16)vv.w;
        *(v4h*)(&Xs[r * XP + c]) = o;
    }
    __syncthreads();

    // ---- Phase 1: S = (X Wq^T + bq)(X Wk^T + bk)^T, accumulated over 6 d-chunks ----
    // Projection mapping: wave w owns COLUMN tile w (weight rows dstart+16w..+15),
    // so each weight fragment is loaded exactly once per workgroup.
    v8f sacc[8];
#pragma unroll
    for (int i = 0; i < 8; ++i) sacc[i] = {};

#pragma unroll 1
    for (int dc = 0; dc < 6; ++dc) {
        const int dstart = dc * 128;

        // ---- Q pass ----
        {
            const float bb = bq[dstart + 16 * w + idx];   // per-column bias
            v8f acc[8];
#pragma unroll
            for (int r = 0; r < 8; ++r)
#pragma unroll
                for (int e = 0; e < 8; ++e) acc[r][e] = bb;   // bias as accumulator init
            const _Float16* wr = WhQ + (size_t)(dstart + 16 * w + idx) * C_;
#pragma unroll 1
            for (int ks = 0; ks < 24; ++ks) {
                const int k0 = ks * 32;
                v16h Bf = load_frag(wr + k0, h);
#pragma unroll
                for (int r = 0; r < 8; ++r) {
                    v16h A = load_frag(&Xs[(16 * r + idx) * XP + k0], h);
                    acc[r] = wmma16(A, Bf, acc[r]);
                }
            }
#pragma unroll
            for (int r = 0; r < 8; ++r)
#pragma unroll
                for (int e = 0; e < 8; ++e)
                    Qs[(16 * r + e + 8 * h) * SP + 16 * w + idx] = (_Float16)acc[r][e];
        }

        // ---- K pass ----
        {
            const float bb = bk[dstart + 16 * w + idx];
            v8f acc[8];
#pragma unroll
            for (int r = 0; r < 8; ++r)
#pragma unroll
                for (int e = 0; e < 8; ++e) acc[r][e] = bb;
            const _Float16* wr = WhK + (size_t)(dstart + 16 * w + idx) * C_;
#pragma unroll 1
            for (int ks = 0; ks < 24; ++ks) {
                const int k0 = ks * 32;
                v16h Bf = load_frag(wr + k0, h);
#pragma unroll
                for (int r = 0; r < 8; ++r) {
                    v16h A = load_frag(&Xs[(16 * r + idx) * XP + k0], h);
                    acc[r] = wmma16(A, Bf, acc[r]);
                }
            }
#pragma unroll
            for (int r = 0; r < 8; ++r)
#pragma unroll
                for (int e = 0; e < 8; ++e)
                    Ks[(16 * r + e + 8 * h) * SP + 16 * w + idx] = (_Float16)acc[r][e];
        }
        __syncthreads();

        // ---- S += Qc * Kc^T ; wave w owns score rows [16w,16w+16); skip kt > w ----
        // A (Q) fragment hoisted: loaded once per dk, reused for all causal kt tiles.
#pragma unroll
        for (int dk = 0; dk < 4; ++dk) {
            const int d0 = dk * 32;
            v16h A = load_frag(&Qs[(16 * w + idx) * SP + d0], h);
#pragma unroll
            for (int kt = 0; kt < 8; ++kt) {
                if (kt <= w) {
                    v16h Bf = load_frag(&Ks[(16 * kt + idx) * SP + d0], h);
                    sacc[kt] = wmma16(A, Bf, sacc[kt]);
                }
            }
        }
        __syncthreads();
    }

    // ---- Phase 2: causal mask + softmax in registers ----
    const float scale = 0.03608439182435161f;  // 1/sqrt(768)
    float rinv[8];
#pragma unroll
    for (int e = 0; e < 8; ++e) {
        const int i_ = 16 * w + e + 8 * h;
        float m = -3.0e38f;
#pragma unroll
        for (int t = 0; t < 8; ++t) {
            const int j = 16 * t + idx;
            float s = (j <= i_) ? sacc[t][e] * scale : -__builtin_inff();
            sacc[t][e] = s;
            m = fmaxf(m, s);
        }
        m = fmaxf(m, __shfl_xor(m, 1, 32));
        m = fmaxf(m, __shfl_xor(m, 2, 32));
        m = fmaxf(m, __shfl_xor(m, 4, 32));
        m = fmaxf(m, __shfl_xor(m, 8, 32));
        float r = 0.0f;
#pragma unroll
        for (int t = 0; t < 8; ++t) {
            float ev = __expf(sacc[t][e] - m);
            sacc[t][e] = ev;
            r += ev;
        }
        r += __shfl_xor(r, 1, 32);
        r += __shfl_xor(r, 2, 32);
        r += __shfl_xor(r, 4, 32);
        r += __shfl_xor(r, 8, 32);
        rinv[e] = __builtin_amdgcn_rcpf(r);
    }
    // store P (f16); each wave writes only its own 16 rows
#pragma unroll
    for (int e = 0; e < 8; ++e) {
#pragma unroll
        for (int t = 0; t < 8; ++t) {
            Ps[(16 * w + e + 8 * h) * SP + 16 * t + idx] = (_Float16)(sacc[t][e] * rinv[e]);
        }
    }

    // ---- Phase 3: O = P * (X Wv^T + bv), chunked over output dim ----
#pragma unroll 1
    for (int dc = 0; dc < 6; ++dc) {
        const int dstart = dc * 128;

        // ---- V pass (column mapping; transposed store VsT[d][j]) ----
        {
            const float bb = bv[dstart + 16 * w + idx];
            v8f acc[8];
#pragma unroll
            for (int r = 0; r < 8; ++r)
#pragma unroll
                for (int e = 0; e < 8; ++e) acc[r][e] = bb;
            const _Float16* wr = WhV + (size_t)(dstart + 16 * w + idx) * C_;
#pragma unroll 1
            for (int ks = 0; ks < 24; ++ks) {
                const int k0 = ks * 32;
                v16h Bf = load_frag(wr + k0, h);
#pragma unroll
                for (int r = 0; r < 8; ++r) {
                    v16h A = load_frag(&Xs[(16 * r + idx) * XP + k0], h);
                    acc[r] = wmma16(A, Bf, acc[r]);
                }
            }
#pragma unroll
            for (int r = 0; r < 8; ++r) {
                v8h pk;
#pragma unroll
                for (int e = 0; e < 8; ++e) pk[e] = (_Float16)acc[r][e];
                // lane's d-row is fixed (16w+idx); j = 16r + 8h + e is contiguous
                *(v8h*)(&VsT[(16 * w + idx) * SP + 16 * r + 8 * h]) = pk;
            }
        }
        __syncthreads();

        // ---- O-chunk = P * Vc ; wave w owns output rows [16w,16w+16) ----
        // P fragments are invariant across output column tiles: preload once.
        {
            v16h Af[4];
#pragma unroll
            for (int kk = 0; kk < 4; ++kk)
                Af[kk] = load_frag(&Ps[(16 * w + idx) * SP + kk * 32], h);
#pragma unroll 1
            for (int t = 0; t < 8; ++t) {
                v8f oa = {};
#pragma unroll
                for (int kk = 0; kk < 4; ++kk) {
                    v16h Bf = load_frag(&VsT[(16 * t + idx) * SP + kk * 32], h);
                    oa = wmma16(Af[kk], Bf, oa);
                }
                float* op = out + ((size_t)b * T_ + 16 * w + 8 * h) * C_ + dstart + 16 * t + idx;
#pragma unroll
                for (int e = 0; e < 8; ++e) op[(size_t)e * C_] = oa[e];
            }
        }
        __syncthreads();
    }
}

extern "C" void kernel_launch(void* const* d_in, const int* in_sizes, int n_in,
                              void* d_out, int out_size, void* d_ws, size_t ws_size,
                              hipStream_t stream) {
    const float* x  = (const float*)d_in[0];
    const float* Wq = (const float*)d_in[1];
    const float* bq = (const float*)d_in[2];
    const float* Wk = (const float*)d_in[3];
    const float* bk = (const float*)d_in[4];
    const float* Wv = (const float*)d_in[5];
    const float* bv = (const float*)d_in[6];
    float* out = (float*)d_out;

    _Float16* wh = (_Float16*)d_ws;  // [Wq_h | Wk_h | Wv_h], 3 * 768*768 f16

    wcvt_kernel<<<(WN + 255) / 256, 256, 0, stream>>>(Wq, Wk, Wv, wh);
    attn_fused_kernel<<<1024, 256, 0, stream>>>(x, wh, wh + WN, wh + 2 * WN,
                                                bq, bk, bv, out);
}
